// DCCF_43344809951756
// MI455X (gfx1250) — matrix operand order repaired and checked by reading, not compiled, not run on hardware
//
#include <hip/hip_runtime.h>
#include <math.h>

#define NU 50000
#define NI 50000
#define NN (NU + NI)
#define DD 32
#define KK 128
#define BB 2048
#define NLAYERS 2
#define NPAIRS 12
#define EMB_REG 2.5e-5f
#define CEN_REG 5e-3f
#define SSL_REG 0.1f
#define EPSV 1e-8f

typedef __attribute__((ext_vector_type(16))) __bf16 v16bf;
typedef __attribute__((ext_vector_type(8)))  float v8f;

#define CDIV(a, b) (((a) + (b) - 1) / (b))

__device__ __forceinline__ float wave_sum(float v) {
#pragma unroll
  for (int m = 16; m >= 1; m >>= 1) v += __shfl_xor(v, m, 32);
  return v;
}

__device__ __forceinline__ unsigned short f2bf(float f) {
  unsigned int u = __float_as_uint(f);
  u += 0x7FFFu + ((u >> 16) & 1u);
  return (unsigned short)(u >> 16);
}

// ---------------------------------------------------------------- utilities
__global__ void fill_kernel(float* p, size_t n, float v) {
  size_t i = (size_t)blockIdx.x * blockDim.x + threadIdx.x;
  if (i < n) p[i] = v;
}

__global__ void init_emb_kernel(const float* ue, const float* ie, float* eCur, float* fin) {
  size_t i = (size_t)blockIdx.x * blockDim.x + threadIdx.x;
  if (i < (size_t)NN * DD) {
    float v = (i < (size_t)NU * DD) ? ue[i] : ie[i - (size_t)NU * DD];
    eCur[i] = v;
    fin[i] = v;
  }
}

// ---------------------------------------------------------------- graph prep
__global__ void degree_kernel(const int* h, float* deg, int E) {
  int e = blockIdx.x * blockDim.x + threadIdx.x;
  if (e < E) atomicAdd(deg + h[e], 1.0f);
}

__global__ void dinv_kernel(float* deg, int n) {
  int i = blockIdx.x * blockDim.x + threadIdx.x;
  if (i < n) {
    float d = deg[i];
    deg[i] = (d > 0.0f) ? (1.0f / sqrtf(d)) : 0.0f;
  }
}

__global__ void gvals_kernel(const float* dinv, const int* h, const int* t, float* g, int E) {
  int e = blockIdx.x * blockDim.x + threadIdx.x;
  if (e < E) g[e] = dinv[h[e]] * dinv[t[e]];
}

// wave per edge is implicit: 32 consecutive threads share edge e, lane = dim
__global__ void spmm_kernel(const float* vals, const int* h, const int* t,
                            const float* x, float* out, int E) {
  size_t idx = (size_t)blockIdx.x * blockDim.x + threadIdx.x;
  int e = (int)(idx >> 5);
  if (e >= E) return;
  int d = (int)(idx & 31);
  float v = vals[e];
  int hh = h[e], tt = t[e];
  atomicAdd(out + (size_t)hh * DD + d, v * x[(size_t)tt * DD + d]);
}

// --------------------------------------------------------------- intent block
__global__ __launch_bounds__(256) void intent_kernel(const float* X, const float* Wu,
                                                     const float* Wi, float* out) {
  __shared__ float sW[2 * DD * KK];  // 32 KB
  for (int i = threadIdx.x; i < DD * KK; i += 256) {
    sW[i] = Wu[i];
    sW[DD * KK + i] = Wi[i];
  }
  __syncthreads();
  int n = blockIdx.x * 256 + threadIdx.x;
  if (n >= NN) return;
  const float* W = (n < NU) ? sW : (sW + DD * KK);
  float e[DD];
#pragma unroll
  for (int d = 0; d < DD; d++) e[d] = X[(size_t)n * DD + d];
  float mx = -3.4e38f;
  for (int k = 0; k < KK; k++) {
    float l = 0.0f;
#pragma unroll
    for (int d = 0; d < DD; d++) l = fmaf(e[d], W[d * KK + k], l);
    mx = fmaxf(mx, l);
  }
  float acc[DD];
#pragma unroll
  for (int d = 0; d < DD; d++) acc[d] = 0.0f;
  float s = 0.0f;
  for (int k = 0; k < KK; k++) {
    float l = 0.0f;
#pragma unroll
    for (int d = 0; d < DD; d++) l = fmaf(e[d], W[d * KK + k], l);
    float pr = __expf(l - mx);
    s += pr;
#pragma unroll
    for (int d = 0; d < DD; d++) acc[d] = fmaf(pr, W[d * KK + k], acc[d]);
  }
  float inv = 1.0f / s;
#pragma unroll
  for (int d = 0; d < DD; d++) out[(size_t)n * DD + d] = acc[d] * inv;
}

// --------------------------------------------------------- adaptive edge vals
__global__ void adaptive_alpha_kernel(const float* X, const int* h, const int* t,
                                      float* alpha, float* rowsum, int E) {
  size_t idx = (size_t)blockIdx.x * blockDim.x + threadIdx.x;
  int e = (int)(idx >> 5);
  if (e >= E) return;
  int lane = threadIdx.x & 31;
  int hh = h[e], tt = t[e];
  float a = X[(size_t)hh * DD + lane];
  float b = X[(size_t)tt * DD + lane];
  float dot = wave_sum(a * b);
  float na = wave_sum(a * a);
  float nb = wave_sum(b * b);
  if (lane == 0) {
    float denom = fmaxf(sqrtf(na), 1e-12f) * fmaxf(sqrtf(nb), 1e-12f);
    float al = (dot / denom + 1.0f) * 0.5f;
    alpha[e] = al;
    atomicAdd(rowsum + hh, al);
  }
}

__global__ void adaptive_vals_kernel(float* vals, const int* h, const float* rowsum, int E) {
  int e = blockIdx.x * blockDim.x + threadIdx.x;
  if (e >= E) return;
  float rs = rowsum[h[e]];
  vals[e] = (rs > 0.0f) ? (vals[e] / rs) : 0.0f;
}

// -------------------------------------------------- slice + l2norm + bf16 pack
__global__ void slice_kernel(const float* gnn, const float* inte, const float* gaa,
                             const float* iaa, const int* users, const int* pitems,
                             unsigned short* slices, int layer) {
  size_t idx = (size_t)blockIdx.x * blockDim.x + threadIdx.x;
  int wid = (int)(idx >> 5);  // 0 .. 8*BB-1
  if (wid >= 8 * BB) return;
  int lane = threadIdx.x & 31;
  int m = wid / BB, r = wid % BB;
  int mm = m & 3;
  const float* src = (mm == 0) ? gnn : (mm == 1) ? inte : (mm == 2) ? gaa : iaa;
  int node = (m < 4) ? users[r] : (NU + pitems[r]);
  float x = src[(size_t)node * DD + lane];
  float n2 = wave_sum(x * x);
  float inv = 1.0f / fmaxf(sqrtf(n2), 1e-12f);
  slices[((size_t)(layer * 8 + m) * BB + r) * DD + lane] = f2bf(x * inv);
}

// -------------------------------------------------------------- layer combine
__global__ void combine_kernel(const float* gnn, const float* inte, const float* gaa,
                               const float* iaa, float* eCur, float* fin) {
  size_t i = (size_t)blockIdx.x * blockDim.x + threadIdx.x;
  if (i < (size_t)NN * DD) {
    float s = gnn[i] + inte[i] + gaa[i] + iaa[i] + eCur[i];
    eCur[i] = s;
    fin[i] += s;
  }
}

// -------------------------------------------------------------------- losses
__global__ void mf_kernel(const float* fin, const int* users, const int* pit,
                          const int* nit, float* acc) {
  size_t idx = (size_t)blockIdx.x * blockDim.x + threadIdx.x;
  int b = (int)(idx >> 5);
  if (b >= BB) return;
  int lane = threadIdx.x & 31;
  const float* u = fin + (size_t)users[b] * DD;
  const float* p = fin + (size_t)(NU + pit[b]) * DD;
  const float* n = fin + (size_t)(NU + nit[b]) * DD;
  float uv = u[lane];
  float dun = wave_sum(uv * n[lane]);
  float dup = wave_sum(uv * p[lane]);
  if (lane == 0) {
    float x = dun - dup;
    float sp = (x > 0.0f) ? (x + log1pf(__expf(-x))) : log1pf(__expf(x));
    atomicAdd(acc, sp);
  }
}

__global__ void embsq_kernel(const float* ue, const float* ie, const int* users,
                             const int* pit, const int* nit, float* acc) {
  size_t idx = (size_t)blockIdx.x * blockDim.x + threadIdx.x;
  float v = 0.0f;
  if (idx < (size_t)3 * BB * DD) {
    int g = (int)(idx / ((size_t)BB * DD));
    int r = (int)((idx / DD) % BB);
    int d = (int)(idx & 31);
    if (g == 0) v = ue[(size_t)users[r] * DD + d];
    else if (g == 1) v = ie[(size_t)pit[r] * DD + d];
    else v = ie[(size_t)nit[r] * DD + d];
  }
  float s = wave_sum(v * v);
  if ((threadIdx.x & 31) == 0) atomicAdd(acc, s);
}

__global__ void censq_kernel(const float* wu, const float* wi, float* acc) {
  size_t idx = (size_t)blockIdx.x * blockDim.x + threadIdx.x;
  float v = 0.0f;
  if (idx < (size_t)2 * DD * KK) {
    v = (idx < (size_t)DD * KK) ? wu[idx] : wi[idx - (size_t)DD * KK];
  }
  float s = wave_sum(v * v);
  if ((threadIdx.x & 31) == 0) atomicAdd(acc, s);
}

// ---------------------------------------------------- contrastive WMMA GEMMs
// One wave per (pair, m-tile). 2048x2048x32 per pair; one v_wmma per 16x16 tile.
// B fragments are double-buffered; the diagonal (pos) is captured under a
// wave-uniform branch and stored once after the loop (no stores in hot loop).
__global__ __launch_bounds__(32) void cl_wmma_kernel(const unsigned short* slicesU,
                                                     float* pos, float* neg) {
  const __bf16* slices = (const __bf16*)slicesU;
  int wid = blockIdx.x;
  int pair = wid >> 7;       // 0..11
  int mt = wid & 127;        // m-tile
  int layer = pair / 6, q = pair % 6;
  int m1 = (q < 3) ? 0 : 4;                 // ug or ig
  int m2 = (q < 3) ? (q + 1) : (q + 2);     // ui/ua2/uia or ii/ia2/iia
  const __bf16* E1 = slices + (size_t)(layer * 8 + m1) * BB * DD;
  const __bf16* E2 = slices + (size_t)(layer * 8 + m2) * BB * DD;

  int lane = threadIdx.x;
  int r16 = lane & 15, half = lane >> 4;

  // A fragment: row = mt*16 + r16; K contiguous half*16 .. half*16+15
  const __bf16* pa = E1 + (size_t)(mt * 16 + r16) * DD + half * 16;
  v16bf a;
#pragma unroll
  for (int i = 0; i < 8; i++) { a[i] = pa[i]; a[i + 8] = pa[i + 8]; }

  float negacc[8];
#pragma unroll
  for (int v = 0; v < 8; v++) negacc[v] = 0.0f;
  float diagval = 0.0f;

  // B fragment base: column = r16, K offset = half*16; tile nt at +nt*16*DD
  const __bf16* pb = E2 + (size_t)r16 * DD + half * 16;
  v16bf bcur;
#pragma unroll
  for (int i = 0; i < 16; i++) bcur[i] = pb[i];

  for (int nt = 0; nt < 128; nt++) {
    // prefetch next B fragment into registers (clamped redundant load on last)
    int ntn = (nt + 1 < 128) ? (nt + 1) : 127;
    const __bf16* pn = pb + (size_t)ntn * 16 * DD;
    v16bf bnxt;
#pragma unroll
    for (int i = 0; i < 16; i++) bnxt[i] = pn[i];

    v8f c = {};
    c = __builtin_amdgcn_wmma_f32_16x16x32_bf16(false, a, false, bcur, (short)0, c,
                                                false, false);
    float ex[8];
#pragma unroll
    for (int v = 0; v < 8; v++) {
      ex[v] = __expf(c[v]);  // TEMP == 1.0
      negacc[v] += ex[v];
    }
    if (nt == mt) {  // wave-uniform scalar branch: diagonal tile, runs once
#pragma unroll
      for (int v = 0; v < 8; v++)
        if (r16 == v + 8 * half) diagval = ex[v];
    }
    bcur = bnxt;
  }

  // pos: lane holds diagonal of row mt*16+r16 iff (r16>>3) == half
  if ((r16 >> 3) == half) pos[(size_t)pair * BB + mt * 16 + r16] = diagval;

  // neg: reduce per-row sums across each 16-lane half (one column slot per lane)
#pragma unroll
  for (int v = 0; v < 8; v++) {
    float s = negacc[v];
#pragma unroll
    for (int m = 8; m >= 1; m >>= 1) s += __shfl_xor(s, m, 32);
    if (r16 == 0) neg[(size_t)pair * BB + mt * 16 + v + half * 8] = s;
  }
}

// ------------------------------------------------------------ final assembly
__global__ void assemble_kernel(const float* pos, const float* neg, const float* accs,
                                float* out) {
  __shared__ float red[256];
  float s = 0.0f;
  for (int i = threadIdx.x; i < NPAIRS * BB; i += 256) {
    s += -logf(pos[i] / (neg[i] + EPSV) + EPSV);
  }
  red[threadIdx.x] = s;
  __syncthreads();
  for (int k = 128; k > 0; k >>= 1) {
    if (threadIdx.x < k) red[threadIdx.x] += red[threadIdx.x + k];
    __syncthreads();
  }
  if (threadIdx.x == 0) {
    out[0] = accs[0] / (float)BB;           // mf_loss
    out[1] = EMB_REG * accs[1];             // emb_loss
    out[2] = CEN_REG * accs[2];             // cen_loss
    out[3] = SSL_REG * (red[0] / (float)BB);// cl_loss
  }
}

// ---------------------------------------------------------------------------
extern "C" void kernel_launch(void* const* d_in, const int* in_sizes, int n_in,
                              void* d_out, int out_size, void* d_ws, size_t ws_size,
                              hipStream_t stream) {
  const float* ue = (const float*)d_in[0];
  const float* ie = (const float*)d_in[1];
  const float* wu = (const float*)d_in[2];
  const float* wi = (const float*)d_in[3];
  const int* all_h = (const int*)d_in[4];
  const int* all_t = (const int*)d_in[5];
  const int* users = (const int*)d_in[6];
  const int* pit = (const int*)d_in[7];
  const int* nit = (const int*)d_in[8];
  int E = in_sizes[4];
  float* out = (float*)d_out;

  // workspace layout (floats)
  float* ws = (float*)d_ws;
  size_t NE = (size_t)NN * DD;
  float* eCur = ws;                       // N x 32 running embedding
  float* fin = eCur + NE;                 // N x 32 final sum
  float* gnn = fin + NE;
  float* inte = gnn + NE;
  float* gaa = inte + NE;
  float* iaa = gaa + NE;
  float* gvals = iaa + NE;                // E
  float* evals = gvals + E;               // E (alpha, then vals in place)
  float* nodebuf = evals + E;             // N (deg/dinv, then rowsum)
  float* pos = nodebuf + NN;              // 12 x 2048
  float* neg = pos + (size_t)NPAIRS * BB; // 12 x 2048
  float* accs = neg + (size_t)NPAIRS * BB;// [mf, emb, cen, pad]
  unsigned short* slices = (unsigned short*)(accs + 8); // 2 x 8 x 2048 x 32 bf16

  const int T = 256;
  // init
  fill_kernel<<<CDIV(NN, T), T, 0, stream>>>(nodebuf, (size_t)NN, 0.0f);
  fill_kernel<<<1, T, 0, stream>>>(accs, (size_t)8, 0.0f);
  init_emb_kernel<<<(int)CDIV(NE, T), T, 0, stream>>>(ue, ie, eCur, fin);

  // normalized adjacency values
  degree_kernel<<<CDIV(E, T), T, 0, stream>>>(all_h, nodebuf, E);
  dinv_kernel<<<CDIV(NN, T), T, 0, stream>>>(nodebuf, NN);
  gvals_kernel<<<CDIV(E, T), T, 0, stream>>>(nodebuf, all_h, all_t, gvals, E);

  int spmmBlocks = (int)CDIV((size_t)E * 32, T);
  for (int L = 0; L < NLAYERS; L++) {
    // gnn = spmm(g_vals, eCur)
    fill_kernel<<<(int)CDIV(NE, T), T, 0, stream>>>(gnn, NE, 0.0f);
    spmm_kernel<<<spmmBlocks, T, 0, stream>>>(gvals, all_h, all_t, eCur, gnn, E);
    // inte = softmax(eCur @ W) @ W.T
    intent_kernel<<<CDIV(NN, T), T, 0, stream>>>(eCur, wu, wi, inte);
    // gaa
    fill_kernel<<<CDIV(NN, T), T, 0, stream>>>(nodebuf, (size_t)NN, 0.0f);
    adaptive_alpha_kernel<<<spmmBlocks, T, 0, stream>>>(gnn, all_h, all_t, evals, nodebuf, E);
    adaptive_vals_kernel<<<CDIV(E, T), T, 0, stream>>>(evals, all_h, nodebuf, E);
    fill_kernel<<<(int)CDIV(NE, T), T, 0, stream>>>(gaa, NE, 0.0f);
    spmm_kernel<<<spmmBlocks, T, 0, stream>>>(evals, all_h, all_t, eCur, gaa, E);
    // iaa
    fill_kernel<<<CDIV(NN, T), T, 0, stream>>>(nodebuf, (size_t)NN, 0.0f);
    adaptive_alpha_kernel<<<spmmBlocks, T, 0, stream>>>(inte, all_h, all_t, evals, nodebuf, E);
    adaptive_vals_kernel<<<CDIV(E, T), T, 0, stream>>>(evals, all_h, nodebuf, E);
    fill_kernel<<<(int)CDIV(NE, T), T, 0, stream>>>(iaa, NE, 0.0f);
    spmm_kernel<<<spmmBlocks, T, 0, stream>>>(evals, all_h, all_t, eCur, iaa, E);
    // extract normalized bf16 slices for contrastive losses
    slice_kernel<<<(int)CDIV((size_t)8 * BB * 32, T), T, 0, stream>>>(
        gnn, inte, gaa, iaa, users, pit, slices, L);
    // eCur += layer terms (in place), fin += new eCur
    combine_kernel<<<(int)CDIV(NE, T), T, 0, stream>>>(gnn, inte, gaa, iaa, eCur, fin);
  }

  // scalar losses
  mf_kernel<<<(int)CDIV((size_t)BB * 32, T), T, 0, stream>>>(fin, users, pit, nit, accs + 0);
  embsq_kernel<<<(int)CDIV((size_t)3 * BB * DD, T), T, 0, stream>>>(ue, ie, users, pit, nit, accs + 1);
  censq_kernel<<<(int)CDIV((size_t)2 * DD * KK, T), T, 0, stream>>>(wu, wi, accs + 2);

  // 12 contrastive GEMMs via WMMA bf16, fused exp/row-sum
  cl_wmma_kernel<<<NPAIRS * 128, 32, 0, stream>>>(slices, pos, neg);

  assemble_kernel<<<1, 256, 0, stream>>>(pos, neg, accs, out);
}
